// Attention_31155692765862
// MI455X (gfx1250) — matrix operand order repaired and checked by reading, not compiled, not run on hardware
//
#include <hip/hip_runtime.h>
#include <hip/hip_bf16.h>

typedef __bf16 bf16;
typedef __attribute__((ext_vector_type(16))) __bf16 v16bf;
typedef __attribute__((ext_vector_type(8)))  __bf16 v8bf;
typedef __attribute__((ext_vector_type(4)))  __bf16 v4bf;
typedef __attribute__((ext_vector_type(2)))  __bf16 v2bf;
typedef __attribute__((ext_vector_type(8)))  float  v8f;
typedef __attribute__((ext_vector_type(4)))  int    v4i;

#define WMMA_BF16(a, b, c) \
  __builtin_amdgcn_wmma_f32_16x16x32_bf16(false, (a), false, (b), (short)0, (c), false, false)

// ---------------- async global->LDS staging (CDNA5 GLOBAL_LOAD_ASYNC_TO_LDS_B128) ----------------
#if __has_builtin(__builtin_amdgcn_global_load_async_to_lds_b128)
#define ASYNC_COPY 1
#else
#define ASYNC_COPY 0
#endif

#if ASYNC_COPY
typedef __attribute__((address_space(1))) v4i* gv4i_p;  // global (device) 128-bit vector ptr
typedef __attribute__((address_space(3))) v4i* lv4i_p;  // LDS 128-bit vector ptr
static __device__ inline void g2l_b128(const bf16* g, bf16* l) {
  __builtin_amdgcn_global_load_async_to_lds_b128((gv4i_p)g, (lv4i_p)l, 0, 0);
}
static __device__ inline void async_wait() {
#if __has_builtin(__builtin_amdgcn_s_wait_asynccnt)
  __builtin_amdgcn_s_wait_asynccnt(0);
#else
  asm volatile("s_wait_asynccnt 0x0" ::: "memory");
#endif
}
#else
static __device__ inline void async_wait() {}
#endif

// Copy a ROWS x (COLS8*8) bf16 tile from global (row stride gs) into LDS (row stride ls).
// NT = threads in block. Fallback path batches 4 global_load_b128 before the ds_stores (MLP=4).
template <int ROWS, int COLS8, int NT>
static __device__ inline void stage_tile(bf16* l, int ls, const bf16* g, size_t gs, int tid) {
  constexpr int IT = (ROWS * COLS8) / NT;
#if ASYNC_COPY
#pragma unroll
  for (int u = 0; u < IT; ++u) {
    int i = tid + u * NT;
    int r = i / COLS8, c = (i % COLS8) * 8;
    g2l_b128(&g[(size_t)r * gs + c], &l[r * ls + c]);
  }
#else
  constexpr int BATCH = IT < 4 ? IT : 4;
#pragma unroll
  for (int base = 0; base < IT; base += BATCH) {
    v8bf tmp[BATCH];
#pragma unroll
    for (int u = 0; u < BATCH; ++u) {
      int i = tid + (base + u) * NT;
      int r = i / COLS8, c = (i % COLS8) * 8;
      tmp[u] = *(const v8bf*)&g[(size_t)r * gs + c];
    }
#pragma unroll
    for (int u = 0; u < BATCH; ++u) {
      int i = tid + (base + u) * NT;
      int r = i / COLS8, c = (i % COLS8) * 8;
      *(v8bf*)&l[r * ls + c] = tmp[u];
    }
  }
#endif
}

// ---------------- WMMA fragment loaders (gfx1250 wave32 layouts) ----------------
static __device__ inline v16bf load_a_frag(const bf16* base, int stride, int lane) {
  int m  = lane & 15;
  int kh = lane >> 4;
  const bf16* r = base + m * stride + kh * 8;
  v8bf lo = *(const v8bf*)r;
  v8bf hi = *(const v8bf*)(r + 16);
  return __builtin_shufflevector(lo, hi, 0, 1, 2, 3, 4, 5, 6, 7, 8, 9, 10, 11, 12, 13, 14, 15);
}

static __device__ inline v16bf load_b_frag_nmaj(const bf16* base, int nstride, int lane) {
  int n  = lane & 15;
  int kh = lane >> 4;
  const bf16* c = base + n * nstride + kh * 8;
  v8bf lo = *(const v8bf*)c;
  v8bf hi = *(const v8bf*)(c + 16);
  return __builtin_shufflevector(lo, hi, 0, 1, 2, 3, 4, 5, 6, 7, 8, 9, 10, 11, 12, 13, 14, 15);
}

// ---------------- generic 64x64 bf16 WMMA GEMM tile (256 threads / 8 waves) ----------------
#define AS_STR 72
#define BS_STR 72

template <int OUTMODE>
static __device__ void gemm64_tile(const bf16* A, int lda, const bf16* Bt, int ldbt,
                                   void* Cv, int ldc, int K, int m0, int n0) {
  __shared__ bf16 As[64 * AS_STR];
  __shared__ bf16 Bs[64 * BS_STR];
  int tid = threadIdx.x;
  int lane = tid & 31, wv = tid >> 5;
  int mi = wv & 3, ni = wv >> 2;
  v8f acc0, acc1;
#pragma unroll
  for (int j = 0; j < 8; ++j) { acc0[j] = 0.f; acc1[j] = 0.f; }

  for (int k0 = 0; k0 < K; k0 += 64) {
    __syncthreads();
    stage_tile<64, 8, 256>(As, AS_STR, &A[(size_t)m0 * lda + k0], (size_t)lda, tid);
    stage_tile<64, 8, 256>(Bs, BS_STR, &Bt[(size_t)n0 * ldbt + k0], (size_t)ldbt, tid);
    async_wait();
    __syncthreads();
#pragma unroll
    for (int kk = 0; kk < 2; ++kk) {
      v16bf a  = load_a_frag(&As[(mi * 16) * AS_STR + kk * 32], AS_STR, lane);
      v16bf b0 = load_b_frag_nmaj(&Bs[(ni * 32) * BS_STR + kk * 32], BS_STR, lane);
      v16bf b1 = load_b_frag_nmaj(&Bs[(ni * 32 + 16) * BS_STR + kk * 32], BS_STR, lane);
      acc0 = WMMA_BF16(a, b0, acc0);
      acc1 = WMMA_BF16(a, b1, acc1);
    }
  }

  int n = lane & 15, mb = (lane >> 4) * 8;
  if (OUTMODE == 2) {
    bf16* C = (bf16*)Cv;  // C^T[(n)*ldc + m], pack row-pairs into 4B stores
#pragma unroll
    for (int v = 0; v < 8; v += 2) {
      size_t mrow = (size_t)(m0 + mi * 16 + mb + v);
      v2bf p0; p0[0] = (bf16)acc0[v]; p0[1] = (bf16)acc0[v + 1];
      v2bf p1; p1[0] = (bf16)acc1[v]; p1[1] = (bf16)acc1[v + 1];
      *(v2bf*)&C[(size_t)(n0 + ni * 32 + n) * ldc + mrow]      = p0;
      *(v2bf*)&C[(size_t)(n0 + ni * 32 + 16 + n) * ldc + mrow] = p1;
    }
  } else {
#pragma unroll
    for (int v = 0; v < 8; ++v) {
      size_t row = (size_t)(m0 + mi * 16 + mb + v);
      if (OUTMODE == 1) {
        bf16* C = (bf16*)Cv;
        C[row * ldc + n0 + ni * 32 + n]      = (bf16)acc0[v];
        C[row * ldc + n0 + ni * 32 + 16 + n] = (bf16)acc1[v];
      } else {
        float* C = (float*)Cv;
        C[row * ldc + n0 + ni * 32 + n]      = acc0[v];
        C[row * ldc + n0 + ni * 32 + 16 + n] = acc1[v];
      }
    }
  }
}

// ---------------- conversion kernels ----------------
__global__ void cvt_bf16_kernel(const float* __restrict__ in, bf16* __restrict__ out, int n) {
  int i = blockIdx.x * blockDim.x + threadIdx.x;
  int stride = gridDim.x * blockDim.x;
  const float4* in4 = (const float4*)in;
  v4bf* out4 = (v4bf*)out;
  for (int i4 = i; i4 < (n >> 2); i4 += stride) {
    float4 f = in4[i4];
    v4bf o;
    o[0] = (bf16)f.x; o[1] = (bf16)f.y; o[2] = (bf16)f.z; o[3] = (bf16)f.w;
    out4[i4] = o;
  }
}

__global__ void cvt_w_t_kernel(const float* __restrict__ in, bf16* __restrict__ out) {
  __shared__ float t[32][33];
  const int E = 512;
  int h = blockIdx.z;
  int e0 = blockIdx.x * 32, f0 = blockIdx.y * 32;
  const float* src = in + (size_t)h * E * E;
  bf16* dst = out + (size_t)h * E * E;
  int tx = threadIdx.x & 31, ty = threadIdx.x >> 5;
  for (int r = ty; r < 32; r += 8)
    t[r][tx] = src[(size_t)(e0 + r) * E + f0 + tx];
  __syncthreads();
  for (int r = ty; r < 32; r += 8)
    dst[(size_t)(f0 + r) * E + e0 + tx] = (bf16)t[tx][r];
}

// ---------------- Q/K/V projections: grid (128 tiles, h*3+j, b) ----------------
__global__ void proj_qkv_kernel(const bf16* __restrict__ xbf, const bf16* __restrict__ wbft,
                                bf16* __restrict__ qk, bf16* __restrict__ vt) {
  const int Sdim = 1024, E = 512, H = 8;
  int hj = blockIdx.y;
  int j = hj % 3, h = hj / 3;
  int b = blockIdx.z;
  int m0 = (blockIdx.x >> 3) * 64;
  int n0 = (blockIdx.x & 7) * 64;
  const bf16* A  = xbf + (size_t)b * Sdim * E;
  const bf16* Bt = wbft + ((size_t)j * H + h) * E * E;
  size_t bh = (size_t)b * H + h;
  if (j == 2) {
    bf16* Ct = vt + bh * (size_t)Sdim * E;  // V stored transposed: (E, S)
    gemm64_tile<2>(A, E, Bt, E, (void*)Ct, Sdim, E, m0, n0);
  } else {
    bf16* C = qk + ((size_t)j * 4 * H + bh) * (size_t)Sdim * E;
    gemm64_tile<1>(A, E, Bt, E, (void*)C, E, E, m0, n0);
  }
}

// q2/k2: one wave per 512-element row, vectorized loads + wave32 shuffle reduction.
__global__ void rownorm_kernel(const bf16* __restrict__ X, float* __restrict__ out, int ncols) {
  int lane = threadIdx.x & 31, wv = threadIdx.x >> 5;
  size_t row = (size_t)blockIdx.x * 8 + wv;
  const v8bf* r8 = (const v8bf*)(X + row * ncols);
  float s = 0.f;
  for (int i = lane; i < (ncols >> 3); i += 32) {
    v8bf v = r8[i];
#pragma unroll
    for (int j = 0; j < 8; ++j) { float f = (float)v[j]; s += f * f; }
  }
#pragma unroll
  for (int off = 16; off > 0; off >>= 1) s += __shfl_down(s, off, 32);
  if (lane == 0) out[row] = s;
}

// ---------------- fused RBF attention ----------------
#define QK_STR 520
#define VS_STR 72
#define PS_STR 72

__global__ void rbf_attn_kernel(const bf16* __restrict__ Q, const bf16* __restrict__ K,
                                const bf16* __restrict__ Vt, const float* __restrict__ q2,
                                const float* __restrict__ k2, const float* __restrict__ gamma,
                                bf16* __restrict__ oh) {
  const int Sdim = 1024, E = 512, H = 8;
  int m0 = blockIdx.x * 64;
  int h = blockIdx.y, b = blockIdx.z;
  int bh = b * H + h;
  const bf16* Qg  = Q  + (size_t)bh * Sdim * E;
  const bf16* Kg  = K  + (size_t)bh * Sdim * E;
  const bf16* Vtg = Vt + (size_t)bh * Sdim * E;  // (E, S)
  const float* q2g = q2 + (size_t)bh * Sdim;
  const float* k2g = k2 + (size_t)bh * Sdim;
  float g = gamma[h];

  extern __shared__ char smem[];
  bf16* QKs = (bf16*)smem;             // 64 x QK_STR (Q at init, K tile in loop)
  bf16* Vs  = QKs + 64 * QK_STR;       // 512 x VS_STR
  bf16* Ps  = Vs + 512 * VS_STR;       // 64 x PS_STR
  float* q2s = (float*)(Ps + 64 * PS_STR);
  float* k2s = q2s + 64;

  int tid = threadIdx.x;
  int lane = tid & 31, wv = tid >> 5;  // 16 waves
  int pm = wv & 3, pn = wv >> 2;       // 4 row-strips x 4 groups

  // stage Q tile, pull per-wave Q fragments into registers, then free the region for K
  stage_tile<64, 64, 512>(QKs, QK_STR, &Qg[(size_t)m0 * E], (size_t)E, tid);
  if (tid < 64) q2s[tid] = q2g[m0 + tid];
  async_wait();
  __syncthreads();

  v16bf qfrag[16];
#pragma unroll
  for (int e = 0; e < 16; ++e)
    qfrag[e] = load_a_frag(&QKs[(pm * 16) * QK_STR + e * 32], QK_STR, lane);

  v8f oacc[8];
#pragma unroll
  for (int t = 0; t < 8; ++t)
#pragma unroll
    for (int j = 0; j < 8; ++j) oacc[t][j] = 0.f;

  __syncthreads();  // all waves done reading Q from LDS

  for (int n0k = 0; n0k < Sdim; n0k += 64) {
    if (n0k + 64 < Sdim)  // global_prefetch for the next key block
      __builtin_prefetch(&Kg[(size_t)(n0k + 64 + (tid & 63)) * E], 0, 1);

    // stage K tile (row-major == n-major for the score GEMM) and V tile
    stage_tile<64, 64, 512>(QKs, QK_STR, &Kg[(size_t)n0k * E], (size_t)E, tid);
    stage_tile<512, 8, 512>(Vs, VS_STR, &Vtg[n0k], (size_t)Sdim, tid);
    if (tid < 64) k2s[tid] = k2g[n0k + tid];
    async_wait();
    __syncthreads();

    // P-phase: 16 WMMAs back-to-back, Q from registers, K from LDS
    v8f pacc;
#pragma unroll
    for (int j = 0; j < 8; ++j) pacc[j] = 0.f;
#pragma unroll
    for (int e = 0; e < 16; ++e) {
      v16bf bB = load_b_frag_nmaj(&QKs[(pn * 16) * QK_STR + e * 32], QK_STR, lane);
      pacc = WMMA_BF16(qfrag[e], bB, pacc);
    }

    // RBF epilogue on f32 accumulator, round to bf16 into LDS P-tile
    {
      int n = lane & 15, mb = (lane >> 4) * 8;
      float k2v = k2s[pn * 16 + n];
#pragma unroll
      for (int v = 0; v < 8; ++v) {
        float d2 = q2s[pm * 16 + mb + v] + k2v - 2.0f * pacc[v];
        d2 = fmaxf(d2, 0.0f);
        float pe = __expf(-g * d2);
        Ps[(pm * 16 + mb + v) * PS_STR + pn * 16 + n] = (bf16)pe;
      }
    }
    __syncthreads();  // Ps visible (V staged earlier in this iteration)

    // O (64x512) += P (64x64) @ V: wave covers rows pm*16.., cols pn*128..
#pragma unroll
    for (int kk = 0; kk < 2; ++kk) {
      v16bf a = load_a_frag(&Ps[(pm * 16) * PS_STR + kk * 32], PS_STR, lane);
#pragma unroll
      for (int t = 0; t < 8; ++t) {
        v16bf bB = load_b_frag_nmaj(&Vs[(pn * 128 + t * 16) * VS_STR + kk * 32], VS_STR, lane);
        oacc[t] = WMMA_BF16(a, bB, oacc[t]);
      }
    }
    __syncthreads();  // protect K/V/P tiles before next key block's staging
  }

  // store concat layout (B, S, H*E) for the output projection
  int n = lane & 15, mb = (lane >> 4) * 8;
#pragma unroll
  for (int t = 0; t < 8; ++t)
#pragma unroll
    for (int v = 0; v < 8; ++v) {
      size_t row = (size_t)m0 + pm * 16 + mb + v;
      oh[((size_t)b * Sdim + row) * (H * E) + h * E + pn * 128 + t * 16 + n] = (bf16)oacc[t][v];
    }
}

// final projection: (4096 x 4096) @ W_o^T -> f32 d_out. W_o native (512,4096) layout IS n-major.
__global__ void out_proj_kernel(const bf16* __restrict__ ohp, const bf16* __restrict__ wobf,
                                float* __restrict__ out) {
  int m0 = (blockIdx.x >> 3) * 64;
  int n0 = (blockIdx.x & 7) * 64;
  gemm64_tile<0>(ohp, 4096, wobf, 4096, (void*)out, 512, 4096, m0, n0);
}

extern "C" void kernel_launch(void* const* d_in, const int* in_sizes, int n_in,
                              void* d_out, int out_size, void* d_ws, size_t ws_size,
                              hipStream_t stream) {
  const int Bb = 4, Sdim = 1024, E = 512, H = 8;
  const float* x     = (const float*)d_in[0];
  const float* Wq    = (const float*)d_in[3];
  const float* Wk    = (const float*)d_in[4];
  const float* Wv    = (const float*)d_in[5];
  const float* Wo    = (const float*)d_in[6];
  const float* gamma = (const float*)d_in[7];
  (void)in_sizes; (void)n_in; (void)out_size;

  char* ws = (char*)d_ws;
  size_t off = 0;
  auto alloc = [&](size_t bytes) -> void* {
    void* p = ws + off;
    off = (off + bytes + 255) & ~(size_t)255;
    return p;
  };

  size_t BHSE = (size_t)Bb * H * Sdim * E;
  bf16*  xbf  = (bf16*)alloc((size_t)Bb * Sdim * E * 2);
  bf16*  wbft = (bf16*)alloc((size_t)3 * H * E * E * 2);
  bf16*  wobf = (bf16*)alloc((size_t)H * E * E * 2);
  bf16*  qk   = (bf16*)alloc(2 * BHSE * 2);
  bf16*  vt   = (bf16*)alloc(BHSE * 2);
  float* q2   = (float*)alloc((size_t)Bb * H * Sdim * 4);
  float* k2   = (float*)alloc((size_t)Bb * H * Sdim * 4);
  bf16*  oh   = (bf16*)alloc((size_t)Bb * Sdim * H * E * 2);
  if (off > ws_size) return;

  int nx = Bb * Sdim * E;
  int nw = H * E * E;
  cvt_bf16_kernel<<<1024, 256, 0, stream>>>(x, xbf, nx);
  cvt_w_t_kernel<<<dim3(16, 16, 8), 256, 0, stream>>>(Wq, wbft);
  cvt_w_t_kernel<<<dim3(16, 16, 8), 256, 0, stream>>>(Wk, wbft + (size_t)nw);
  cvt_w_t_kernel<<<dim3(16, 16, 8), 256, 0, stream>>>(Wv, wbft + (size_t)2 * nw);
  cvt_bf16_kernel<<<1024, 256, 0, stream>>>(Wo, wobf, nw);

  proj_qkv_kernel<<<dim3(128, 24, 4), 256, 0, stream>>>(xbf, wbft, qk, vt);

  bf16* Qb = qk;
  bf16* Kb = qk + BHSE;
  rownorm_kernel<<<4096, 256, 0, stream>>>(Qb, q2, E);
  rownorm_kernel<<<4096, 256, 0, stream>>>(Kb, k2, E);

  size_t shmem = (size_t)(64 * QK_STR + 512 * VS_STR + 64 * PS_STR) * 2 + 128 * 4;
  (void)hipFuncSetAttribute((const void*)rbf_attn_kernel,
                            hipFuncAttributeMaxDynamicSharedMemorySize, (int)shmem);
  rbf_attn_kernel<<<dim3(16, 8, 4), 512, shmem, stream>>>(Qb, Kb, vt, q2, k2, gamma, oh);

  out_proj_kernel<<<512, 256, 0, stream>>>(oh, wobf, (float*)d_out);
}
